// Block_69999376990592
// MI455X (gfx1250) — matrix operand order repaired and checked by reading, not compiled, not run on hardware
//
#include <hip/hip_runtime.h>
#include <hip/hip_bf16.h>
#include <math.h>

// ---------------------------------------------------------------------------
// Transformer block for MI455X (gfx1250, wave32, WMMA).
// - GEMMs: v_wmma_f32_16x16x32_bf16, 128x128 block tile, double-buffered LDS
//   staged with GLOBAL_LOAD_ASYNC_TO_LDS_B128 (ASYNCcnt-tracked).
// - Weights pre-transposed to [N][K] bf16 so all staging is contiguous.
// - Attention: 32x2048 fp32 score tile resident in 320KB WGP LDS.
// ---------------------------------------------------------------------------

typedef __attribute__((ext_vector_type(16))) __bf16 v16bf;
typedef __attribute__((ext_vector_type(8)))  __bf16 v8bf;
typedef __attribute__((ext_vector_type(8)))  float  v8f;

#define DMODEL 1024
#define NTOK   8192      // B*N = 4*2048
#define SEQ    2048
#define NHEAD  16
#define HDIM   64

static __device__ __forceinline__ v8f wmma_bf16(const v16bf& a, const v16bf& b, const v8f& c) {
    return __builtin_amdgcn_wmma_f32_16x16x32_bf16(
        /*neg_a=*/false, a, /*neg_b=*/false, b,
        /*c_mod=*/(short)0, c, /*reuse_a=*/false, /*reuse_b=*/false);
}

// Async global->LDS copy of 16 bytes (one lane): CDNA5 GLOBAL_LOAD_ASYNC_TO_LDS_B128.
static __device__ __forceinline__ void async_copy_b128(unsigned lds_off, const void* gptr) {
    unsigned long long ga = (unsigned long long)gptr;
    asm volatile("global_load_async_to_lds_b128 %0, %1, off"
                 :: "v"(lds_off), "v"(ga) : "memory");
}
static __device__ __forceinline__ void wait_async0() {
    asm volatile("s_wait_asynccnt 0" ::: "memory");
}
static __device__ __forceinline__ unsigned lds_off_of(const void* p) {
    return (unsigned)(size_t)p;   // low 32 bits of generic ptr = LDS offset
}

// ---------------------------------------------------------------------------
// fp32 [R][C] -> bf16 transposed [C][R]. 32x32 LDS tile, 256 threads (32x8).
// ---------------------------------------------------------------------------
__global__ __launch_bounds__(256)
void cast_transpose_bf16_kernel(const float* __restrict__ in, __bf16* __restrict__ out,
                                int R, int C) {
    __shared__ __bf16 tile[32][33];
    const int bx = blockIdx.x * 32;           // col base (C dim)
    const int by = blockIdx.y * 32;           // row base (R dim)
    const int tx = threadIdx.x & 31, ty = threadIdx.x >> 5;
    #pragma unroll
    for (int i = 0; i < 32; i += 8)
        tile[ty + i][tx] = (__bf16)in[(size_t)(by + ty + i) * C + bx + tx];
    __syncthreads();
    #pragma unroll
    for (int i = 0; i < 32; i += 8)
        out[(size_t)(bx + ty + i) * R + by + tx] = tile[tx][ty + i];
}

// ---------------------------------------------------------------------------
// LayerNorm over D=1024, fp32 in -> bf16 out. One 256-thread block per row.
// ---------------------------------------------------------------------------
__global__ __launch_bounds__(256) void layernorm_bf16_kernel(const float* __restrict__ x,
                                                             const float* __restrict__ g,
                                                             const float* __restrict__ b,
                                                             __bf16* __restrict__ out) {
    const int row  = blockIdx.x;
    const int tid  = threadIdx.x;
    const int lane = tid & 31, wid = tid >> 5;
    const float* xr = x + (size_t)row * DMODEL;

    float4 v = *(const float4*)&xr[tid * 4];
    float s  = v.x + v.y + v.z + v.w;
    float s2 = v.x * v.x + v.y * v.y + v.z * v.z + v.w * v.w;
    #pragma unroll
    for (int o = 16; o > 0; o >>= 1) {
        s  += __shfl_xor(s, o, 32);
        s2 += __shfl_xor(s2, o, 32);
    }
    __shared__ float ps[8], ps2[8];
    if (lane == 0) { ps[wid] = s; ps2[wid] = s2; }
    __syncthreads();
    float tot = 0.f, tot2 = 0.f;
    #pragma unroll
    for (int i = 0; i < 8; ++i) { tot += ps[i]; tot2 += ps2[i]; }
    const float mu  = tot * (1.0f / DMODEL);
    const float var = tot2 * (1.0f / DMODEL) - mu * mu;
    const float inv = rsqrtf(var + 1e-5f);

    __bf16* orow = out + (size_t)row * DMODEL;
    const int c0 = tid * 4;
    float vv[4] = {v.x, v.y, v.z, v.w};
    #pragma unroll
    for (int e = 0; e < 4; ++e) {
        int c = c0 + e;
        orow[c] = (__bf16)((vv[e] - mu) * inv * g[c] + b[c]);
    }
}

// ---------------------------------------------------------------------------
// bf16 WMMA GEMM: C[M,N] = act(A[M,K] * Bt^T + bias) (+res), Bt is [N][K].
// 256 threads = 8 waves. Block tile 128x128, BK=32, wave tile 64x32 (4x2 frags).
// Double-buffered LDS, staged with async global->LDS copies.
// ---------------------------------------------------------------------------
template <bool BIAS, bool GELU_ACT, bool RES, bool OUTBF>
__global__ __launch_bounds__(256)
void gemm_bf16_kernel(const __bf16* __restrict__ A, const __bf16* __restrict__ Bt,
                      const float* __restrict__ bias, const float* __restrict__ res,
                      float* __restrict__ Cf, __bf16* __restrict__ Cb,
                      int M, int N, int K) {
    constexpr int BM = 128, BN = 128, BK = 32;
    __shared__ __bf16 As[2][BM][BK];   // 2 x 8 KB
    __shared__ __bf16 Bs[2][BN][BK];   // 2 x 8 KB  ([n][k] -> frag loads contiguous)

    const int tid  = threadIdx.x;
    const int lane = tid & 31, wid = tid >> 5;
    const int bm = blockIdx.y * BM, bn = blockIdx.x * BN;
    const int wm = (wid & 1) * 64, wn = (wid >> 1) * 32;

    const int nl = lane & 15;            // m (A) / n (B/C) lane index
    const int hi = lane >> 4;            // half-wave select
    const int kA = hi * 16;              // A-frag k base   (contiguous k = 16*hi + s, matches B)
    const int kB = hi * 16;              // B-frag k base   (contiguous k = 16*hi + s, matches A)

    v8f acc[4][2] = {};

    // Async stage of one K-tile (16 KB): 4 x b128 per thread.
    auto stage = [&](int buf, int k0) {
        #pragma unroll
        for (int it = 0; it < 2; ++it) {
            const int c = tid + it * 256;
            const int r = c >> 2, cc = (c & 3) * 8;
            async_copy_b128(lds_off_of(&As[buf][r][cc]),
                            &A[(size_t)(bm + r) * K + k0 + cc]);
            async_copy_b128(lds_off_of(&Bs[buf][r][cc]),
                            &Bt[(size_t)(bn + r) * K + k0 + cc]);
        }
    };

    stage(0, 0);
    wait_async0();
    __syncthreads();

    const int nk = K / BK;
    for (int kt = 0; kt < nk; ++kt) {
        const int buf = kt & 1;
        if (kt + 1 < nk) stage(buf ^ 1, (kt + 1) * BK);   // prefetch next tile

        v16bf bfr[2];
        #pragma unroll
        for (int j = 0; j < 2; ++j)
            bfr[j] = *(const v16bf*)&Bs[buf][wn + j * 16 + nl][kB];
        #pragma unroll
        for (int i = 0; i < 4; ++i) {
            const __bf16* ap = &As[buf][wm + i * 16 + nl][0];
            v8bf alo = *(const v8bf*)&ap[kA];
            v8bf ahi = *(const v8bf*)&ap[kA + 8];
            v16bf a;
            #pragma unroll
            for (int j = 0; j < 8; ++j) { a[j] = alo[j]; a[8 + j] = ahi[j]; }
            #pragma unroll
            for (int j = 0; j < 2; ++j)
                acc[i][j] = wmma_bf16(a, bfr[j], acc[i][j]);
        }

        wait_async0();      // next tile fully in LDS
        __syncthreads();    // all waves done reading current tile
    }

    // Epilogue: C row = base + r + 8*hi (ISA C/D layout), col = base + nl.
    #pragma unroll
    for (int i = 0; i < 4; ++i) {
        #pragma unroll
        for (int j = 0; j < 2; ++j) {
            #pragma unroll
            for (int r = 0; r < 8; ++r) {
                const int row = bm + wm + i * 16 + hi * 8 + r;
                const int col = bn + wn + j * 16 + nl;
                float v = acc[i][j][r];
                if constexpr (BIAS) v += bias[col];
                if constexpr (GELU_ACT) v = 0.5f * v * (1.0f + erff(v * 0.70710678118f));
                if constexpr (RES) v += res[(size_t)row * N + col];
                if constexpr (OUTBF) Cb[(size_t)row * N + col] = (__bf16)v;
                else                 Cf[(size_t)row * N + col] = v;
            }
        }
    }
}

// ---------------------------------------------------------------------------
// Attention: one workgroup per (b, h, 32-query tile). Full 32x2048 fp32 score
// tile lives in LDS (256 KB of the 320 KB WGP LDS). qkv is bf16 [8192, 3072]
// (cols 0..1023 = Q, 1024..2047 = K, 2048..3071 = V; head h at h*64).
// Output O written bf16 [8192, 1024].
// ---------------------------------------------------------------------------
__global__ __launch_bounds__(256)
void attention_kernel(const __bf16* __restrict__ qkv, __bf16* __restrict__ o) {
    extern __shared__ char smem[];
    float*  Sf  = (float*)smem;                          // [32][2048] fp32
    __bf16* Qs  = (__bf16*)(smem + 32 * SEQ * 4);        // [32][64]
    __bf16* VsT = Qs + 32 * HDIM;                        // [64][32] transposed V tile

    const int tid  = threadIdx.x;
    const int lane = tid & 31, wid = tid >> 5;
    const int bh = blockIdx.x >> 6;       // 64 query tiles per (b,h)
    const int qt = blockIdx.x & 63;
    const int b  = bh >> 4, h = bh & 15;
    const size_t base = (size_t)b * SEQ * (3 * DMODEL);
    const int q0 = qt * 32;

    const int nl = lane & 15;
    const int hi = lane >> 4;
    const int kA = hi * 16;
    const int kB = hi * 16;
    const int msub = (wid & 1) * 16;

    // Stage Q tile (32x64 bf16) via async copy.
    {
        int r = tid >> 3, c = (tid & 7) * 8;
        async_copy_b128(lds_off_of(&Qs[r * HDIM + c]),
                        &qkv[base + (size_t)(q0 + r) * (3 * DMODEL) + h * HDIM + c]);
    }
    wait_async0();
    __syncthreads();

    // Phase 1: S = (Q * K^T) / sqrt(64). 2x128 tiles of 16x16 across 8 waves.
    for (int ntb = 0; ntb < 32; ++ntb) {
        const int nt   = (wid >> 1) + ntb * 4;   // 0..127
        const int key0 = nt * 16;
        v8f acc = {};
        #pragma unroll
        for (int kk0 = 0; kk0 < HDIM; kk0 += 32) {
            const __bf16* ap = &Qs[(msub + nl) * HDIM + kk0];
            v8bf alo = *(const v8bf*)&ap[kA];
            v8bf ahi = *(const v8bf*)&ap[kA + 8];
            v16bf a;
            #pragma unroll
            for (int j = 0; j < 8; ++j) { a[j] = alo[j]; a[8 + j] = ahi[j]; }
            // B frag: K rows are contiguous in head-dim -> single 32B load/lane.
            v16bf bf = *(const v16bf*)&qkv[base + (size_t)(key0 + nl) * (3 * DMODEL)
                                           + DMODEL + h * HDIM + kk0 + kB];
            acc = wmma_bf16(a, bf, acc);
        }
        #pragma unroll
        for (int r = 0; r < 8; ++r) {
            const int rr = msub + hi * 8 + r;
            Sf[rr * SEQ + key0 + nl] = acc[r] * 0.125f;   // 1/sqrt(64)
        }
    }
    __syncthreads();

    // Phase 2: softmax rows in LDS. Each wave owns 4 rows; wave32 shuffles.
    for (int ri = 0; ri < 4; ++ri) {
        float* row = &Sf[(wid * 4 + ri) * SEQ];
        float m = -3.402823466e38f;
        for (int c = lane; c < SEQ; c += 32) m = fmaxf(m, row[c]);
        #pragma unroll
        for (int o2 = 16; o2 > 0; o2 >>= 1) m = fmaxf(m, __shfl_xor(m, o2, 32));
        float ssum = 0.f;
        for (int c = lane; c < SEQ; c += 32) {
            float e = __expf(row[c] - m);
            row[c] = e;
            ssum += e;
        }
        #pragma unroll
        for (int o2 = 16; o2 > 0; o2 >>= 1) ssum += __shfl_xor(ssum, o2, 32);
        const float inv = 1.0f / ssum;
        for (int c = lane; c < SEQ; c += 32) row[c] *= inv;
    }
    __syncthreads();

    // Phase 3: O = P(32x2048) * V(2048x64). 8 output tiles across 8 waves.
    const int nsub = (wid >> 1) * 16;     // 0,16,32,48 within head dim
    v8f acc = {};
    for (int kt = 0; kt < SEQ / 32; ++kt) {
        const int k0 = kt * 32;
        // Cooperatively stage V tile transposed: VsT[vd][k].
        {
            int kk = tid >> 3, vb = (tid & 7) * 8;
            const __bf16* vp = &qkv[base + (size_t)(k0 + kk) * (3 * DMODEL)
                                    + 2 * DMODEL + h * HDIM + vb];
            #pragma unroll
            for (int j = 0; j < 8; ++j) VsT[(vb + j) * 32 + kk] = vp[j];
        }
        __syncthreads();
        // A frag: fp32 P from LDS, convert to bf16 at load.
        const float* sp = &Sf[(msub + nl) * SEQ + k0];
        v16bf a;
        #pragma unroll
        for (int j = 0; j < 8; ++j) a[j] = (__bf16)sp[kA + j];
        #pragma unroll
        for (int j = 0; j < 8; ++j) a[8 + j] = (__bf16)sp[kA + 8 + j];
        v16bf bf = *(const v16bf*)&VsT[(nsub + nl) * 32 + kB];
        acc = wmma_bf16(a, bf, acc);
        __syncthreads();
    }
    #pragma unroll
    for (int r = 0; r < 8; ++r) {
        const int rr = msub + hi * 8 + r;
        o[(size_t)(b * SEQ + q0 + rr) * DMODEL + h * HDIM + nsub + nl] = (__bf16)acc[r];
    }
}

// ---------------------------------------------------------------------------
// Launcher
// ---------------------------------------------------------------------------
extern "C" void kernel_launch(void* const* d_in, const int* in_sizes, int n_in,
                              void* d_out, int out_size, void* d_ws, size_t ws_size,
                              hipStream_t stream) {
    const float* x      = (const float*)d_in[0];
    const float* w_qkv  = (const float*)d_in[1];
    const float* w_proj = (const float*)d_in[2];
    const float* b_proj = (const float*)d_in[3];
    const float* w1     = (const float*)d_in[4];
    const float* b1     = (const float*)d_in[5];
    const float* w2     = (const float*)d_in[6];
    const float* b2     = (const float*)d_in[7];
    const float* g1     = (const float*)d_in[8];
    const float* be1    = (const float*)d_in[9];
    const float* g2     = (const float*)d_in[10];
    const float* be2    = (const float*)d_in[11];
    float* out = (float*)d_out;

    // Workspace carve-up (256B aligned), ~216 MB total.
    char* ws = (char*)d_ws;
    size_t off = 0;
    auto carve = [&](size_t bytes) -> char* {
        char* p = ws + off;
        off += (bytes + 255) & ~(size_t)255;
        return p;
    };
    __bf16* wqkvT  = (__bf16*)carve((size_t)DMODEL * 3 * DMODEL * 2);  // [3072][1024]
    __bf16* wprojT = (__bf16*)carve((size_t)DMODEL * DMODEL * 2);      // [1024][1024]
    __bf16* w1T    = (__bf16*)carve((size_t)DMODEL * 4 * DMODEL * 2);  // [4096][1024]
    __bf16* w2T    = (__bf16*)carve((size_t)4 * DMODEL * DMODEL * 2);  // [1024][4096]
    __bf16* ln1_bf = (__bf16*)carve((size_t)NTOK * DMODEL * 2);
    __bf16* qkv_bf = (__bf16*)carve((size_t)NTOK * 3 * DMODEL * 2);
    __bf16* attn_bf= (__bf16*)carve((size_t)NTOK * DMODEL * 2);
    float*  x1     = (float*)carve((size_t)NTOK * DMODEL * 4);
    __bf16* h_bf   = (__bf16*)carve((size_t)NTOK * DMODEL * 2);
    __bf16* h1_bf  = (__bf16*)carve((size_t)NTOK * 4 * DMODEL * 2);

    // 1) Weights -> bf16, transposed to [N][K].
    cast_transpose_bf16_kernel<<<dim3(3 * DMODEL / 32, DMODEL / 32), 256, 0, stream>>>(
        w_qkv, wqkvT, DMODEL, 3 * DMODEL);
    cast_transpose_bf16_kernel<<<dim3(DMODEL / 32, DMODEL / 32), 256, 0, stream>>>(
        w_proj, wprojT, DMODEL, DMODEL);
    cast_transpose_bf16_kernel<<<dim3(4 * DMODEL / 32, DMODEL / 32), 256, 0, stream>>>(
        w1, w1T, DMODEL, 4 * DMODEL);
    cast_transpose_bf16_kernel<<<dim3(DMODEL / 32, 4 * DMODEL / 32), 256, 0, stream>>>(
        w2, w2T, 4 * DMODEL, DMODEL);

    // 2) ln1 = LN(x, g1, be1) -> bf16.
    layernorm_bf16_kernel<<<NTOK, 256, 0, stream>>>(x, g1, be1, ln1_bf);

    // 3) qkv = ln1 @ w_qkv (bf16 out, no bias).
    gemm_bf16_kernel<false, false, false, true><<<dim3(3 * DMODEL / 128, NTOK / 128),
        256, 0, stream>>>(ln1_bf, wqkvT, nullptr, nullptr, nullptr, qkv_bf,
                          NTOK, 3 * DMODEL, DMODEL);

    // 4) Attention -> attn_bf. 4096 blocks, 264 KB LDS each.
    const size_t attn_smem = (size_t)32 * SEQ * 4 + 32 * HDIM * 2 + HDIM * 32 * 2;
    attention_kernel<<<4 * NHEAD * (SEQ / 32), 256, attn_smem, stream>>>(qkv_bf, attn_bf);

    // 5) x1 = x + attn @ w_proj + b_proj (fp32 out).
    gemm_bf16_kernel<true, false, true, false><<<dim3(DMODEL / 128, NTOK / 128),
        256, 0, stream>>>(attn_bf, wprojT, b_proj, x, x1, nullptr,
                          NTOK, DMODEL, DMODEL);

    // 6) h = LN(x1, g2, be2) -> bf16.
    layernorm_bf16_kernel<<<NTOK, 256, 0, stream>>>(x1, g2, be2, h_bf);

    // 7) h1 = gelu(h @ w1 + b1) -> bf16.
    gemm_bf16_kernel<true, true, false, true><<<dim3(4 * DMODEL / 128, NTOK / 128),
        256, 0, stream>>>(h_bf, w1T, b1, nullptr, nullptr, h1_bf,
                          NTOK, 4 * DMODEL, DMODEL);

    // 8) out = x1 + h1 @ w2 + b2 (fp32 out).
    gemm_bf16_kernel<true, false, true, false><<<dim3(DMODEL / 128, NTOK / 128),
        256, 0, stream>>>(h1_bf, w2T, b2, x1, out, nullptr,
                          NTOK, DMODEL, 4 * DMODEL);
}
